// Structure2Vec_8993661518205
// MI455X (gfx1250) — matrix operand order repaired and checked by reading, not compiled, not run on hardware
//
#include <hip/hip_runtime.h>
#include <hip/hip_bf16.h>

// Problem constants (from reference)
#define N_NODES   100000
#define N_EDGES   1600000
#define FEAT      128        // F_IN == H == 128
#define C_OUT     10
#define G_GRAPHS  64

typedef __attribute__((ext_vector_type(16))) __bf16 v16bf;
typedef __attribute__((ext_vector_type(8)))  float  v8f;

// -------------------------------------------------------------------------
// Weight prep: W [128x128] row-major fp32 -> WT [n][k] bf16 (transposed)
// so the WMMA B-fragment (lane = column n, 16 contiguous K per half-wave)
// is a single aligned 32-byte vector load.
// -------------------------------------------------------------------------
__global__ void wt_prep_kernel(const float* __restrict__ W,
                               __bf16* __restrict__ WT) {
    int idx = blockIdx.x * blockDim.x + threadIdx.x;
    if (idx >= FEAT * FEAT) return;
    int n = idx >> 7, k = idx & 127;
    WT[n * FEAT + k] = (__bf16)W[k * FEAT + n];
}

// -------------------------------------------------------------------------
// GEMM: Y[N,128] = A[N,128] @ W  via v_wmma_f32_16x16x32_bf16.
// A is fp32; converted to bf16 fragments in-register. If BIAS_RELU,
// A is first mapped through relu(A + bias[k]) (fuses layer-1 epilogue
// into layer-2's operand load -> saves a full 51MB pass).
// One wave computes a 16x128 output slab: 4 A-fragments, 32 WMMAs.
// -------------------------------------------------------------------------
template <bool BIAS_RELU>
__global__ void gemm_bf16_kernel(const float* __restrict__ A,
                                 const __bf16* __restrict__ WT,
                                 const float* __restrict__ bias,
                                 float* __restrict__ Y,
                                 int nRowTiles) {
    int wave = blockIdx.x * (blockDim.x >> 5) + (threadIdx.x >> 5);
    if (wave >= nRowTiles) return;          // wave-uniform exit (EXEC all-1 for WMMA)
    int lane = threadIdx.x & 31;
    int r    = lane & 15;
    int half = lane >> 4;

    const float* arow = A + (long)(wave * 16 + r) * FEAT;

    // A fragments: lane holds row r; element e -> K = k0+16*half+e
    // (matches B). Four contiguous float4 loads.
    v16bf afrag[4];
#pragma unroll
    for (int kt = 0; kt < 4; ++kt) {
        int c0 = kt * 32 + 16 * half;
        float4 f0 = *(const float4*)(arow + c0);
        float4 f1 = *(const float4*)(arow + c0 + 4);
        float4 f2 = *(const float4*)(arow + c0 + 8);
        float4 f3 = *(const float4*)(arow + c0 + 12);
        float v[16] = {f0.x, f0.y, f0.z, f0.w, f1.x, f1.y, f1.z, f1.w,
                       f2.x, f2.y, f2.z, f2.w, f3.x, f3.y, f3.z, f3.w};
        if (BIAS_RELU) {
#pragma unroll
            for (int e = 0; e < 16; ++e) {
                int feat = c0 + e;
                float t = v[e] + bias[feat];
                v[e] = t > 0.f ? t : 0.f;
            }
        }
#pragma unroll
        for (int e = 0; e < 16; ++e) afrag[kt][e] = (__bf16)v[e];
    }

    // 8 column tiles of 16; B fragment: lane = col, K = k0 + 16*half + e.
#pragma unroll
    for (int nt = 0; nt < 8; ++nt) {
        v8f acc = {};
        const __bf16* bcol = WT + (nt * 16 + r) * FEAT + 16 * half;
#pragma unroll
        for (int kt = 0; kt < 4; ++kt) {
            v16bf bfrag = *(const v16bf*)(bcol + kt * 32);
            acc = __builtin_amdgcn_wmma_f32_16x16x32_bf16(
                false, afrag[kt], false, bfrag, (short)0, acc, false, false);
        }
        // D layout: VGPR v -> row v + 8*half, lane&15 -> column.
#pragma unroll
        for (int v = 0; v < 8; ++v)
            Y[(long)(wave * 16 + v + 8 * half) * FEAT + nt * 16 + r] = acc[v];
    }
}

// -------------------------------------------------------------------------
// Degree / normalization
// -------------------------------------------------------------------------
__global__ void deg_init_kernel(float* __restrict__ deg) {
    int i = blockIdx.x * blockDim.x + threadIdx.x;
    if (i < N_NODES) deg[i] = 1.0f;               // self-loop
}
__global__ void deg_edge_kernel(const int* __restrict__ dst,
                                float* __restrict__ deg, int nE) {
    int i = blockIdx.x * blockDim.x + threadIdx.x;
    if (i < nE) atomicAdd(&deg[dst[i]], 1.0f);
}
__global__ void deg_finalize_kernel(float* __restrict__ deg) {
    int i = blockIdx.x * blockDim.x + threadIdx.x;
    if (i >= N_NODES) return;
    float d = deg[i];
    deg[i] = rsqrtf(d > 1.0f ? d : 1.0f);         // deg -> dis in place
}

// out[i,:] = xw[i,:] * dis[i]^2   (self-loop contribution, initializes agg)
__global__ void selfloop_kernel(const float* __restrict__ xw,
                                const float* __restrict__ dis,
                                float* __restrict__ out) {
    int idx = blockIdx.x * blockDim.x + threadIdx.x;   // N * 32 threads
    if (idx >= N_NODES * 32) return;
    int node = idx >> 5, q = (idx & 31) * 4;
    float s = dis[node];
    float n2 = s * s;
    float4 v = *(const float4*)(xw + (long)node * FEAT + q);
    float4 o = {v.x * n2, v.y * n2, v.z * n2, v.w * n2};
    *(float4*)(out + (long)node * FEAT + q) = o;
}

// One wave per edge: float4 gather of xw[src], scaled atomic scatter to out[dst].
// Working set (51MB) fits in the 192MB L2 -> L2-atomic bound, not HBM bound.
__global__ void scatter_edge_kernel(const int* __restrict__ src,
                                    const int* __restrict__ dst,
                                    const float* __restrict__ dis,
                                    const float* __restrict__ xw,
                                    float* __restrict__ out, int nE) {
    int e = blockIdx.x * (blockDim.x >> 5) + (threadIdx.x >> 5);
    if (e >= nE) return;                          // wave-uniform
    int s = src[e], d = dst[e];
    float nrm = dis[s] * dis[d];
    int q = (threadIdx.x & 31) * 4;
    float4 v = *(const float4*)(xw + (long)s * FEAT + q);
    float* o = out + (long)d * FEAT + q;
    atomicAdd(o + 0, v.x * nrm);
    atomicAdd(o + 1, v.y * nrm);
    atomicAdd(o + 2, v.z * nrm);
    atomicAdd(o + 3, v.w * nrm);
}

// -------------------------------------------------------------------------
// Pool: relu(agg + b2) segment-summed into [G,128] + counts.
// 128 threads/block; thread f exclusively owns LDS column f of all 64 graph
// rows -> conflict-free accumulation, then 64 global atomics per thread.
// -------------------------------------------------------------------------
__global__ void pool_kernel(const float* __restrict__ agg,
                            const float* __restrict__ b2,
                            const int* __restrict__ batch,
                            float* __restrict__ pool,   // [G,128]
                            float* __restrict__ cnt,    // [G]
                            int nodesPerBlock) {
    __shared__ float acc[G_GRAPHS * FEAT];
    __shared__ float cl[G_GRAPHS];
    int f = threadIdx.x;  // 0..127
    for (int g = 0; g < G_GRAPHS; ++g) acc[g * FEAT + f] = 0.f;
    if (f < G_GRAPHS) cl[f] = 0.f;
    __syncthreads();
    float bias = b2[f];
    int start = blockIdx.x * nodesPerBlock;
    int end   = start + nodesPerBlock;
    if (end > N_NODES) end = N_NODES;
    for (int i = start; i < end; ++i) {
        int g = batch[i];
        float v = agg[(long)i * FEAT + f] + bias;
        acc[g * FEAT + f] += (v > 0.f ? v : 0.f);
        if (f == 0) cl[g] += 1.0f;
    }
    __syncthreads();
    for (int g = 0; g < G_GRAPHS; ++g) atomicAdd(&pool[g * FEAT + f], acc[g * FEAT + f]);
    if (f < G_GRAPHS) atomicAdd(&cnt[f], cl[f]);
}

__global__ void zero_kernel(float* __restrict__ p, int n) {
    int i = blockIdx.x * blockDim.x + threadIdx.x;
    if (i < n) p[i] = 0.f;
}

// out[g,c] = (pool[g,:] @ Wfc[:,c]) / max(cnt[g],1) + bfc[c]   (tiny, 640 outputs)
__global__ void head_kernel(const float* __restrict__ pool,
                            const float* __restrict__ cnt,
                            const float* __restrict__ Wfc,
                            const float* __restrict__ bfc,
                            float* __restrict__ out) {
    int idx = blockIdx.x * blockDim.x + threadIdx.x;
    if (idx >= G_GRAPHS * C_OUT) return;
    int g = idx / C_OUT, c = idx % C_OUT;
    float cv = cnt[g];
    float inv = 1.0f / (cv > 1.0f ? cv : 1.0f);
    float s = 0.f;
#pragma unroll 8
    for (int h = 0; h < FEAT; ++h) s += pool[g * FEAT + h] * Wfc[h * C_OUT + c];
    out[idx] = s * inv + bfc[c];
}

// -------------------------------------------------------------------------
extern "C" void kernel_launch(void* const* d_in, const int* in_sizes, int n_in,
                              void* d_out, int out_size, void* d_ws, size_t ws_size,
                              hipStream_t stream) {
    const float* x    = (const float*)d_in[0];
    const int*   eidx = (const int*)  d_in[1];
    const int*   batch= (const int*)  d_in[2];
    const float* W1   = (const float*)d_in[3];
    const float* b1   = (const float*)d_in[4];
    const float* W2   = (const float*)d_in[5];
    const float* b2   = (const float*)d_in[6];
    const float* Wfc  = (const float*)d_in[7];
    const float* bfc  = (const float*)d_in[8];
    float* out = (float*)d_out;

    const int* src = eidx;             // edge_index[0]
    const int* dst = eidx + N_EDGES;   // edge_index[1]

    // Workspace carve-out (~103 MB total)
    char* ws = (char*)d_ws;
    size_t off = 0;
    auto carve = [&](size_t bytes) -> void* {
        void* p = ws + off;
        off = (off + bytes + 255) & ~(size_t)255;
        return p;
    };
    float*  xw   = (float*) carve((size_t)N_NODES * FEAT * 4); // gemm out (reused for layer 2)
    float*  agg  = (float*) carve((size_t)N_NODES * FEAT * 4); // scatter accumulator (reused)
    float*  dis  = (float*) carve((size_t)N_NODES * 4);        // deg -> rsqrt in place
    __bf16* WT1  = (__bf16*)carve((size_t)FEAT * FEAT * 2);
    __bf16* WT2  = (__bf16*)carve((size_t)FEAT * FEAT * 2);
    float*  pool = (float*) carve((size_t)(G_GRAPHS * FEAT + G_GRAPHS) * 4);
    float*  cnt  = pool + G_GRAPHS * FEAT;

    const int nRowTiles = N_NODES / 16;                  // 6250 exactly
    const int gemmBlocks = (nRowTiles + 7) / 8;          // 8 waves per 256-thr block
    const int edgeBlocks = (N_EDGES + 7) / 8;            // 8 edges per block

    // Weight prep
    wt_prep_kernel<<<(FEAT * FEAT + 255) / 256, 256, 0, stream>>>(W1, WT1);
    wt_prep_kernel<<<(FEAT * FEAT + 255) / 256, 256, 0, stream>>>(W2, WT2);

    // Degree -> dis
    deg_init_kernel<<<(N_NODES + 255) / 256, 256, 0, stream>>>(dis);
    deg_edge_kernel<<<(N_EDGES + 255) / 256, 256, 0, stream>>>(dst, dis, N_EDGES);
    deg_finalize_kernel<<<(N_NODES + 255) / 256, 256, 0, stream>>>(dis);

    // Layer 1: xw = x @ W1 ; agg = scatter(xw * norm)
    gemm_bf16_kernel<false><<<gemmBlocks, 256, 0, stream>>>(x, WT1, nullptr, xw, nRowTiles);
    selfloop_kernel<<<(N_NODES * 32 + 255) / 256, 256, 0, stream>>>(xw, dis, agg);
    scatter_edge_kernel<<<edgeBlocks, 256, 0, stream>>>(src, dst, dis, xw, agg, N_EDGES);

    // Layer 2: xw2 = relu(agg + b1) @ W2 (epilogue fused into A-load); agg = scatter
    gemm_bf16_kernel<true><<<gemmBlocks, 256, 0, stream>>>(agg, WT2, b1, xw, nRowTiles);
    selfloop_kernel<<<(N_NODES * 32 + 255) / 256, 256, 0, stream>>>(xw, dis, agg);
    scatter_edge_kernel<<<edgeBlocks, 256, 0, stream>>>(src, dst, dis, xw, agg, N_EDGES);

    // Pool (bias+relu fused) and head
    zero_kernel<<<(G_GRAPHS * FEAT + G_GRAPHS + 255) / 256, 256, 0, stream>>>(
        pool, G_GRAPHS * FEAT + G_GRAPHS);
    const int nodesPerBlock = 512;
    pool_kernel<<<(N_NODES + nodesPerBlock - 1) / nodesPerBlock, FEAT, 0, stream>>>(
        agg, b2, batch, pool, cnt, nodesPerBlock);
    head_kernel<<<(G_GRAPHS * C_OUT + 255) / 256, 256, 0, stream>>>(pool, cnt, Wfc, bfc, out);
}
